// PinSAGEModel_23278722744485
// MI455X (gfx1250) — compile-verified, matrix operand order checked by reading, not run on hardware
//
#include <hip/hip_runtime.h>
#include <hip/hip_bf16.h>

// ---------------------------------------------------------------------------
// PinSAGE pipeline for MI455X (gfx1250, wave32).
// GEMMs use v_wmma_f32_16x16x32_bf16 (bf16 operands, f32 accumulate).
// Segment sums use hardware f32 global atomics into an L2-resident buffer.
// ---------------------------------------------------------------------------

typedef __attribute__((ext_vector_type(16))) __bf16 v16bf;
typedef __attribute__((ext_vector_type(8)))  float  v8f;

#define N_SRC   131072
#define N_MID   32768
#define N_DST   8192
#define E1_N    524288
#define E2_N    131072
#define HDIM    256
#define N_PAIRS 8192

// ---------------------------------------------------------------- utilities
__device__ __forceinline__ void atomic_add_f32(float* p, float v) {
    unsafeAtomicAdd(p, v);   // hardware global_atomic_add_f32
}

__global__ void f32_to_bf16_kernel(const float* __restrict__ src,
                                   __bf16* __restrict__ dst, int n) {
    int i = blockIdx.x * blockDim.x + threadIdx.x;
    if (i < n) dst[i] = (__bf16)src[i];
}

__global__ void zero_f32_kernel(float* __restrict__ p, int n) {
    int i = blockIdx.x * blockDim.x + threadIdx.x;
    if (i < n) p[i] = 0.0f;
}

// ------------------------------------------------------------------- GEMM
// C[M,256] = act(A[M,K] @ W[256,K]^T + bias), W pre-converted to bf16.
// One wave computes a 16x64 strip: A fragment loaded once per K-step,
// reused across 4 WMMA B tiles. Block = 8 waves = 32 rows x 256 cols.
__global__ __launch_bounds__(256)
void gemm_bf16_n256(const float* __restrict__ A, const __bf16* __restrict__ W,
                    const float* __restrict__ bias, float* __restrict__ C,
                    int M, int K, int relu)
{
    const int lane = threadIdx.x & 31;
    const int wave = threadIdx.x >> 5;
    const int half = lane >> 4;       // K-half selector (ISA 16-bit layout)
    const int l15  = lane & 15;       // row (A) / col (B) within tile
    const int m0   = blockIdx.x * 32 + (wave >> 2) * 16;
    const int n0   = (wave & 3) * 64;

    v8f acc[4];
    #pragma unroll
    for (int t = 0; t < 4; ++t) acc[t] = (v8f){};

    const float* arow = A + (size_t)(m0 + l15) * K;

    for (int k0 = 0; k0 < K; k0 += 32) {
        // A fragment: 16x32 bf16, lane = M row, packed pairs along K.
        v16bf a;
        #pragma unroll
        for (int q = 0; q < 8; ++q) {
            int kk = k0 + (q >> 2) * 16 + half * 8 + (q & 3) * 2;
            float2 v = *(const float2*)(arow + kk);
            a[2 * q]     = (__bf16)v.x;
            a[2 * q + 1] = (__bf16)v.y;
        }
        #pragma unroll
        for (int t = 0; t < 4; ++t) {
            // B fragment: column n of B == row n of row-major W.
            const __bf16* wrow = W + (size_t)(n0 + t * 16 + l15) * K;
            v16bf b;
            #pragma unroll
            for (int q = 0; q < 8; ++q) {
                int kk = k0 + (q >> 2) * 16 + half * 8 + (q & 3) * 2;
                b[2 * q]     = wrow[kk];
                b[2 * q + 1] = wrow[kk + 1];
            }
            acc[t] = __builtin_amdgcn_wmma_f32_16x16x32_bf16(
                false, a, false, b, (short)0, acc[t], false, false);
        }
    }

    // D layout: VGPR r -> M = r + 8*half, N = lane&15.
    #pragma unroll
    for (int t = 0; t < 4; ++t) {
        int n = n0 + t * 16 + l15;
        float bv = bias[n];
        #pragma unroll
        for (int r = 0; r < 8; ++r) {
            int m = m0 + half * 8 + r;
            float v = acc[t][r] + bv;
            if (relu) v = fmaxf(v, 0.0f);
            C[(size_t)m * HDIM + n] = v;
        }
    }
}

// --------------------------------------------- edge gather-mul-scatter-add
// One wave per edge: m_e = n[src_e] * w_e, scatter-added into agg[dst_e].
__global__ __launch_bounds__(256)
void edge_scatter_kernel(const float* __restrict__ nsrc,
                         const float* __restrict__ w,
                         const int* __restrict__ src,
                         const int* __restrict__ dst,
                         float* __restrict__ agg,
                         float* __restrict__ wsum, int E)
{
    int e    = blockIdx.x * 8 + (threadIdx.x >> 5);
    int lane = threadIdx.x & 31;
    if (e >= E) return;
    int   s  = src[e];
    int   d  = dst[e];
    float we = w[e];
    const float* nr = nsrc + (size_t)s * HDIM + lane * 8;
    float*       ar = agg  + (size_t)d * HDIM + lane * 8;
    float4 v0 = *(const float4*)(nr);
    float4 v1 = *(const float4*)(nr + 4);
    atomic_add_f32(ar + 0, v0.x * we);
    atomic_add_f32(ar + 1, v0.y * we);
    atomic_add_f32(ar + 2, v0.z * we);
    atomic_add_f32(ar + 3, v0.w * we);
    atomic_add_f32(ar + 4, v1.x * we);
    atomic_add_f32(ar + 5, v1.y * we);
    atomic_add_f32(ar + 6, v1.z * we);
    atomic_add_f32(ar + 7, v1.w * we);
    if (lane == 0) atomic_add_f32(wsum + d, we);
}

// --------------------------------------- z = concat(agg / clip(ws,1), hdst)
__global__ void build_z_kernel(const float* __restrict__ agg,
                               const float* __restrict__ wsum,
                               const float* __restrict__ hdst,
                               float* __restrict__ z, int rows)
{
    int i = blockIdx.x * blockDim.x + threadIdx.x;
    if (i >= rows * 512) return;
    int r = i >> 9, c = i & 511;
    float v;
    if (c < HDIM) {
        float ws = wsum[r];
        ws = ws < 1.0f ? 1.0f : ws;
        v = agg[(size_t)r * HDIM + c] * __frcp_rn(ws);
    } else {
        v = hdst[(size_t)r * HDIM + (c - HDIM)];
    }
    z[i] = v;
}

// ------------------------------------- z = l2norm(hp[:N_DST] + h2), per row
__global__ __launch_bounds__(256)
void skip_norm_kernel(const float* __restrict__ hp, const float* __restrict__ h2,
                      float* __restrict__ zf, int rows)
{
    int r    = blockIdx.x * 8 + (threadIdx.x >> 5);
    int lane = threadIdx.x & 31;
    if (r >= rows) return;
    float v[8];
    float ss = 0.0f;
    #pragma unroll
    for (int j = 0; j < 8; ++j) {
        int c = lane * 8 + j;
        v[j] = hp[(size_t)r * HDIM + c] + h2[(size_t)r * HDIM + c];
        ss += v[j] * v[j];
    }
    #pragma unroll
    for (int o = 16; o > 0; o >>= 1) ss += __shfl_xor(ss, o, 32);
    float nrm   = sqrtf(ss);
    float scale = (nrm == 0.0f) ? 1.0f : 1.0f / nrm;
    #pragma unroll
    for (int j = 0; j < 8; ++j)
        zf[(size_t)r * HDIM + lane * 8 + j] = v[j] * scale;
}

// --------------------------------------------------- margin scoring kernel
__global__ __launch_bounds__(256)
void score_kernel(const float* __restrict__ zf, const float* __restrict__ bias,
                  const int* __restrict__ nids,
                  const int* __restrict__ ps, const int* __restrict__ pd,
                  const int* __restrict__ ns, const int* __restrict__ nd,
                  float* __restrict__ out, int npairs)
{
    int p    = blockIdx.x * 8 + (threadIdx.x >> 5);
    int lane = threadIdx.x & 31;
    if (p >= npairs) return;
    int pu = ps[p], pv = pd[p], nu = ns[p], nv = nd[p];
    float sp = 0.0f, sn = 0.0f;
    #pragma unroll
    for (int j = 0; j < 8; ++j) {
        int c = lane * 8 + j;
        sp += zf[(size_t)pu * HDIM + c] * zf[(size_t)pv * HDIM + c];
        sn += zf[(size_t)nu * HDIM + c] * zf[(size_t)nv * HDIM + c];
    }
    #pragma unroll
    for (int o = 16; o > 0; o >>= 1) {
        sp += __shfl_xor(sp, o, 32);
        sn += __shfl_xor(sn, o, 32);
    }
    if (lane == 0) {
        float gp = bias[nids[pu]] + bias[nids[pv]];
        float gn = bias[nids[nu]] + bias[nids[nv]];
        float m  = (sn + gn) - (sp + gp) + 1.0f;
        out[p] = m > 0.0f ? m : 0.0f;
    }
}

// ---------------------------------------------------------------------------
extern "C" void kernel_launch(void* const* d_in, const int* in_sizes, int n_in,
                              void* d_out, int out_size, void* d_ws, size_t ws_size,
                              hipStream_t stream) {
    const float* feat = (const float*)d_in[0];
    const float* Wp   = (const float*)d_in[1];
    const float* bp   = (const float*)d_in[2];
    const float* Q1   = (const float*)d_in[3];
    const float* bq1  = (const float*)d_in[4];
    const float* W1   = (const float*)d_in[5];
    const float* bw1  = (const float*)d_in[6];
    const float* Q2   = (const float*)d_in[7];
    const float* bq2  = (const float*)d_in[8];
    const float* W2   = (const float*)d_in[9];
    const float* bw2  = (const float*)d_in[10];
    const float* w1v  = (const float*)d_in[11];
    const float* w2v  = (const float*)d_in[12];
    const float* bias = (const float*)d_in[13];
    const int* src1   = (const int*)d_in[14];
    const int* dst1   = (const int*)d_in[15];
    const int* src2   = (const int*)d_in[16];
    const int* dst2   = (const int*)d_in[17];
    const int* pos_s  = (const int*)d_in[18];
    const int* pos_d  = (const int*)d_in[19];
    const int* neg_s  = (const int*)d_in[20];
    const int* neg_d  = (const int*)d_in[21];
    const int* nids   = (const int*)d_in[22];
    float* out        = (float*)d_out;

    // -------- workspace layout (bytes) --------
    char* ws = (char*)d_ws;
    const size_t SZ_HP  = (size_t)N_SRC * HDIM * 4;      // 134 MB
    const size_t SZ_N   = (size_t)N_SRC * HDIM * 4;      // 134 MB (n1/z1/n2/z2)
    const size_t SZ_H1  = (size_t)N_MID * HDIM * 4;      // 33 MB
    const size_t SZ_AGG = (size_t)N_MID * HDIM * 4;      // 33 MB
    const size_t SZ_WS  = 256 * 1024;                    // wsum (padded)
    const size_t SZ_H2  = (size_t)N_DST * HDIM * 4;      // 8 MB
    const size_t SZ_ZF  = (size_t)N_DST * HDIM * 4;      // 8 MB
    float*  hp   = (float*)(ws);
    float*  nbuf = (float*)(ws + SZ_HP);
    float*  h1   = (float*)(ws + SZ_HP + SZ_N);
    float*  agg  = (float*)(ws + SZ_HP + SZ_N + SZ_H1);
    float*  wsum = (float*)(ws + SZ_HP + SZ_N + SZ_H1 + SZ_AGG);
    float*  h2   = (float*)(ws + SZ_HP + SZ_N + SZ_H1 + SZ_AGG + SZ_WS);
    float*  zf   = (float*)(ws + SZ_HP + SZ_N + SZ_H1 + SZ_AGG + SZ_WS + SZ_H2);
    __bf16* Wpb  = (__bf16*)(ws + SZ_HP + SZ_N + SZ_H1 + SZ_AGG + SZ_WS + SZ_H2 + SZ_ZF);
    __bf16* Q1b  = Wpb + HDIM * HDIM;
    __bf16* W1b  = Q1b + HDIM * HDIM;
    __bf16* Q2b  = W1b + HDIM * 2 * HDIM;
    __bf16* W2b  = Q2b + HDIM * HDIM;

    // 1) weights -> bf16 (stay hot in L2 for all GEMMs)
    f32_to_bf16_kernel<<<(HDIM * HDIM + 255) / 256, 256, 0, stream>>>(Wp, Wpb, HDIM * HDIM);
    f32_to_bf16_kernel<<<(HDIM * HDIM + 255) / 256, 256, 0, stream>>>(Q1, Q1b, HDIM * HDIM);
    f32_to_bf16_kernel<<<(HDIM * 2 * HDIM + 255) / 256, 256, 0, stream>>>(W1, W1b, HDIM * 2 * HDIM);
    f32_to_bf16_kernel<<<(HDIM * HDIM + 255) / 256, 256, 0, stream>>>(Q2, Q2b, HDIM * HDIM);
    f32_to_bf16_kernel<<<(HDIM * 2 * HDIM + 255) / 256, 256, 0, stream>>>(W2, W2b, HDIM * 2 * HDIM);

    // 2) hp = feat @ Wp^T + bp
    gemm_bf16_n256<<<N_SRC / 32, 256, 0, stream>>>(feat, Wpb, bp, hp, N_SRC, HDIM, 0);
    // 3) n1 = relu(hp @ Q1^T + bq1)
    gemm_bf16_n256<<<N_SRC / 32, 256, 0, stream>>>(hp, Q1b, bq1, nbuf, N_SRC, HDIM, 1);
    // 4) layer-1 segment sums
    {
        int nz = N_MID * HDIM + 65536;   // agg + wsum (contiguous)
        zero_f32_kernel<<<(nz + 255) / 256, 256, 0, stream>>>(agg, nz);
        edge_scatter_kernel<<<E1_N / 8, 256, 0, stream>>>(nbuf, w1v, src1, dst1, agg, wsum, E1_N);
    }
    // 5) z1 = [agg/ws, hp[:N_MID]] (reuse nbuf), h1 = relu(z1 @ W1^T + bw1)
    build_z_kernel<<<(N_MID * 512 + 255) / 256, 256, 0, stream>>>(agg, wsum, hp, nbuf, N_MID);
    gemm_bf16_n256<<<N_MID / 32, 256, 0, stream>>>(nbuf, W1b, bw1, h1, N_MID, 2 * HDIM, 1);
    // 6) n2 = relu(h1 @ Q2^T + bq2) (reuse nbuf)
    gemm_bf16_n256<<<N_MID / 32, 256, 0, stream>>>(h1, Q2b, bq2, nbuf, N_MID, HDIM, 1);
    // 7) layer-2 segment sums
    {
        int nz = N_DST * HDIM;
        zero_f32_kernel<<<(nz + 255) / 256, 256, 0, stream>>>(agg, nz);
        zero_f32_kernel<<<(N_DST + 255) / 256, 256, 0, stream>>>(wsum, N_DST);
        edge_scatter_kernel<<<E2_N / 8, 256, 0, stream>>>(nbuf, w2v, src2, dst2, agg, wsum, E2_N);
    }
    // 8) z2 = [agg/ws, h1[:N_DST]], h2 = relu(z2 @ W2^T + bw2)
    build_z_kernel<<<(N_DST * 512 + 255) / 256, 256, 0, stream>>>(agg, wsum, h1, nbuf, N_DST);
    gemm_bf16_n256<<<N_DST / 32, 256, 0, stream>>>(nbuf, W2b, bw2, h2, N_DST, 2 * HDIM, 1);
    // 9) z = l2norm(hp[:N_DST] + h2)
    skip_norm_kernel<<<N_DST / 8, 256, 0, stream>>>(hp, h2, zf, N_DST);
    // 10) margin scores
    score_kernel<<<N_PAIRS / 8, 256, 0, stream>>>(zf, bias, nids,
                                                  pos_s, pos_d, neg_s, neg_d,
                                                  out, N_PAIRS);
}
